// ListREPLValueHead_22643067584656
// MI455X (gfx1250) — matrix operand order repaired and checked by reading, not compiled, not run on hardware
//
#include <hip/hip_runtime.h>
#include <math.h>

typedef __attribute__((ext_vector_type(16))) __bf16 v16bf;
typedef __attribute__((ext_vector_type(8)))  float  v8f;

#define NTILES 8    // N is always 512 -> 8 tiles of 64 columns

// ---------------- helpers ----------------

__device__ __forceinline__ __bf16 f2bf(float f) {
    union { float f; unsigned u; } in; in.f = f;
    unsigned u = in.u;
    unsigned r = u + 0x7FFFu + ((u >> 16) & 1u);     // round-to-nearest-even
    union { unsigned short s; __bf16 b; } out;
    out.s = (unsigned short)(r >> 16);
    return out.b;
}

// A fragment (16x32 bf16, 16-bit A layout): lane half=0 -> K {0..7, 16..23},
// half=1 -> K {8..15, 24..31}.  p already includes row base + half*8.
__device__ __forceinline__ v16bf load_a_frag(const __bf16* p) {
    union { uint4 u[2]; v16bf v; } t;
    t.u[0] = *(const uint4*)(p);
    t.u[1] = *(const uint4*)(p + 16);
    return t.v;
}

// B fragment (32x16 bf16): lane half=0 -> K {0..7, 16..23},
// half=1 -> K {8..15, 24..31}.  p already includes col base + half*8.
__device__ __forceinline__ v16bf load_b_frag(const __bf16* p) {
    union { uint4 u[2]; v16bf v; } t;
    t.u[0] = *(const uint4*)(p);
    t.u[1] = *(const uint4*)(p + 16);
    return t.v;
}

struct Frags { v16bf a0, a1, b0, b1, b2, b3; };

__device__ __forceinline__ Frags load_frags(
    const __bf16* a0, const __bf16* a1,
    const __bf16* bp0, const __bf16* bp1,
    const __bf16* bp2, const __bf16* bp3, int kb)
{
    Frags f;
    f.a0 = load_a_frag(a0 + kb);
    f.a1 = load_a_frag(a1 + kb);
    f.b0 = load_b_frag(bp0 + kb);
    f.b1 = load_b_frag(bp1 + kb);
    f.b2 = load_b_frag(bp2 + kb);
    f.b3 = load_b_frag(bp3 + kb);
    return f;
}

__device__ __forceinline__ void mma8(const Frags& f, v8f (&acc)[2][4]) {
    acc[0][0] = __builtin_amdgcn_wmma_f32_16x16x32_bf16(false, f.a0, false, f.b0, (short)0, acc[0][0], false, false);
    acc[0][1] = __builtin_amdgcn_wmma_f32_16x16x32_bf16(false, f.a0, false, f.b1, (short)0, acc[0][1], false, false);
    acc[0][2] = __builtin_amdgcn_wmma_f32_16x16x32_bf16(false, f.a0, false, f.b2, (short)0, acc[0][2], false, false);
    acc[0][3] = __builtin_amdgcn_wmma_f32_16x16x32_bf16(false, f.a0, false, f.b3, (short)0, acc[0][3], false, false);
    acc[1][0] = __builtin_amdgcn_wmma_f32_16x16x32_bf16(false, f.a1, false, f.b0, (short)0, acc[1][0], false, false);
    acc[1][1] = __builtin_amdgcn_wmma_f32_16x16x32_bf16(false, f.a1, false, f.b1, (short)0, acc[1][1], false, false);
    acc[1][2] = __builtin_amdgcn_wmma_f32_16x16x32_bf16(false, f.a1, false, f.b2, (short)0, acc[1][2], false, false);
    acc[1][3] = __builtin_amdgcn_wmma_f32_16x16x32_bf16(false, f.a1, false, f.b3, (short)0, acc[1][3], false, false);
}

// Double-buffered pipelined K sub-loop over [kBeg, kEnd); A pointers fixed
// for the whole range (no per-iteration source select).
__device__ __forceinline__ void gemm_kloop(
    const __bf16* a0, const __bf16* a1,
    const __bf16* bp0, const __bf16* bp1,
    const __bf16* bp2, const __bf16* bp3,
    int kBeg, int kEnd, v8f (&acc)[2][4])
{
    Frags f = load_frags(a0, a1, bp0, bp1, bp2, bp3, kBeg);
#pragma unroll 2
    for (int kb = kBeg + 32; kb < kEnd; kb += 32) {
        Frags fn = load_frags(a0, a1, bp0, bp1, bp2, bp3, kb);
        mma8(f, acc);
        f = fn;
    }
    mma8(f, acc);
}

// ---------------- WMMA GEMM ----------------
// C[M x 512] = A[M x K] * WT^T (+bias), 32x64 C tile per wave (8 WMMAs/step).
// Dual-source A handled as two pipelined phases:
//   phase 0: k in [0, kSplit)  from Alo (row stride ldaLo)
//   phase 1: k in [kSplit, K)  from Ahi (row stride ldaHi), pointer pre-shifted by -kSplit
// op: 0=none, 1=relu, 2=tanh.  Optional fp32 store (Cf) and bf16 store (Cb).
__global__ void __launch_bounds__(256) gemm_wmma_bf16(
    const __bf16* __restrict__ Alo, int ldaLo,
    const __bf16* __restrict__ Ahi, int ldaHi, int kSplit,
    const __bf16* __restrict__ WT, int K,
    int Mtiles,
    const float* __restrict__ bias,
    float* __restrict__ Cf, int ldcf,
    __bf16* __restrict__ Cb, int ldcb,
    int op)
{
    const int tid  = threadIdx.x;
    const int wave = tid >> 5;
    const int lane = tid & 31;
    const int tile = blockIdx.x * 8 + wave;
    if (tile >= Mtiles * NTILES) return;
    const int mt   = tile / NTILES;
    const int nt   = tile % NTILES;
    const int half = lane >> 4;
    const int l15  = lane & 15;
    const int aOff = half * 8;
    const int bOff = half * 8;

    const int m0 = mt * 32 + l15;
    const __bf16* a0Lo = Alo + (size_t)m0        * ldaLo + aOff;
    const __bf16* a1Lo = Alo + (size_t)(m0 + 16) * ldaLo + aOff;

    const int n0 = nt * 64 + l15;
    const __bf16* bp0 = WT + (size_t)(n0     ) * K + bOff;
    const __bf16* bp1 = WT + (size_t)(n0 + 16) * K + bOff;
    const __bf16* bp2 = WT + (size_t)(n0 + 32) * K + bOff;
    const __bf16* bp3 = WT + (size_t)(n0 + 48) * K + bOff;

    v8f acc[2][4];
#pragma unroll
    for (int i = 0; i < 2; ++i)
#pragma unroll
        for (int j = 0; j < 4; ++j) acc[i][j] = {};

    // phase 0: lo source, k in [0, kSplit)
    gemm_kloop(a0Lo, a1Lo, bp0, bp1, bp2, bp3, 0, kSplit, acc);

    // phase 1: hi source, k in [kSplit, K)   (zero trips for single-source GEMMs)
    if (Ahi) {
        const __bf16* a0Hi = Ahi + (size_t)m0        * ldaHi + aOff - kSplit;
        const __bf16* a1Hi = Ahi + (size_t)(m0 + 16) * ldaHi + aOff - kSplit;
        gemm_kloop(a0Hi, a1Hi, bp0, bp1, bp2, bp3, kSplit, K, acc);
    }

    // fused epilogue
#pragma unroll
    for (int i = 0; i < 2; ++i) {
#pragma unroll
        for (int j = 0; j < 4; ++j) {
            const int n  = n0 + j * 16;
            const float bj = bias ? bias[n] : 0.f;
#pragma unroll
            for (int r = 0; r < 8; ++r) {
                const int m = mt * 32 + i * 16 + r + half * 8;  // C layout: VGPR r -> M=r (+8 hi half)
                float v = acc[i][j][r] + bj;
                if (op == 1)      v = fmaxf(v, 0.f);
                else if (op == 2) v = tanhf(v);
                if (Cf) Cf[(size_t)m * ldcf + n] = v;
                if (Cb) Cb[(size_t)m * ldcb + n] = f2bf(v);
            }
        }
    }
}

// ---------------- elementwise kernels ----------------

// dst[n*dstLd + k] = bf16(src[k*N + n])  -- N-major bf16 weight panels
__global__ void transpose_f32_to_bf16(__bf16* __restrict__ dst, int dstLd,
                                      const float* __restrict__ src, int K, int N) {
    int idx = blockIdx.x * blockDim.x + threadIdx.x;
    if (idx >= K * N) return;
    int k = idx / N, n = idx % N;
    dst[(size_t)n * dstLd + k] = f2bf(src[idx]);
}

__global__ void cast_f32_bf16(const float* __restrict__ src, __bf16* __restrict__ dst, int n) {
    int idx = blockIdx.x * blockDim.x + threadIdx.x;
    if (idx < n) dst[idx] = f2bf(src[idx]);
}

__global__ void zero_u32(unsigned* __restrict__ p, int n) {
    int idx = blockIdx.x * blockDim.x + threadIdx.x;
    if (idx < n) p[idx] = 0u;
}

// leaf[(be*16 + l)*512 + h] = relu(xh[be,h] + emb[leaf_ids[b,l],h] + b_hole[h])
__global__ void leaf_build(const float* __restrict__ xh, const int* __restrict__ leaf_ids,
                           const float* __restrict__ emb, const float* __restrict__ b_hole,
                           __bf16* __restrict__ leaf) {
    int idx = blockIdx.x * blockDim.x + threadIdx.x;   // 4096*16*512 = 33.5M
    if (idx >= 4096 * 16 * 512) return;
    int h  = idx & 511;
    int l  = (idx >> 9) & 15;
    int be = idx >> 13;
    int b  = be >> 3;
    float v = xh[(size_t)be * 512 + h]
            + emb[(size_t)leaf_ids[b * 16 + l] * 512 + h]
            + b_hole[h];
    leaf[idx] = f2bf(fmaxf(v, 0.f));
}

// compared[b, c] = max_e concat(cmp, ctx)[b, e, c]   (stored bf16, 512x1024)
__global__ void compared_max_kernel(const float* __restrict__ cmp, const float* __restrict__ ctx,
                                    __bf16* __restrict__ out) {
    int idx = blockIdx.x * blockDim.x + threadIdx.x;
    if (idx >= 512 * 1024) return;
    int b = idx >> 10, c = idx & 1023;
    const float* src = (c < 512) ? cmp : ctx;
    int cc = c & 511;
    float m = -INFINITY;
#pragma unroll
    for (int e = 0; e < 8; ++e)
        m = fmaxf(m, src[((size_t)(b * 8 + e)) * 512 + cc]);
    out[idx] = f2bf(m);
}

// out[row] = softplus(dot(o1[row,:], W_o2) + b_o2)  -- one wave per row
__global__ void final_head(const float* __restrict__ o1, const float* __restrict__ W_o2,
                           const float* __restrict__ b_o2, float* __restrict__ out) {
    int wave = threadIdx.x >> 5, lane = threadIdx.x & 31;
    int row = blockIdx.x * 8 + wave;
    if (row >= 512) return;
    const float* r = o1 + (size_t)row * 512;
    float s = 0.f;
    for (int i = lane; i < 512; i += 32) s += r[i] * W_o2[i];
#pragma unroll
    for (int off = 16; off > 0; off >>= 1) s += __shfl_down(s, off, 32);
    if (lane == 0) {
        float z = s + b_o2[0];
        out[row] = (z > 0.f) ? z + log1pf(expf(-z)) : log1pf(expf(z));
    }
}

// ---------------- host ----------------

extern "C" void kernel_launch(void* const* d_in, const int* in_sizes, int n_in,
                              void* d_out, int out_size, void* d_ws, size_t ws_size,
                              hipStream_t stream) {
    (void)in_sizes; (void)n_in; (void)out_size; (void)ws_size;

    const float* x        = (const float*)d_in[0];
    const int*   leaf_ids = (const int*)  d_in[1];
    const float* emb      = (const float*)d_in[2];
    const float* W_hole   = (const float*)d_in[3];
    const float* b_hole   = (const float*)d_in[4];
    const float* W_fn     = (const float*)d_in[5];
    const float* b_fn     = (const float*)d_in[6];
    const float* W_ih     = (const float*)d_in[7];
    const float* W_hh     = (const float*)d_in[8];
    const float* b_rnn    = (const float*)d_in[9];
    const float* W_cmp    = (const float*)d_in[10];
    const float* b_cmp    = (const float*)d_in[11];
    const float* W_o1     = (const float*)d_in[12];
    const float* b_o1     = (const float*)d_in[13];
    const float* W_o2     = (const float*)d_in[14];
    const float* b_o2     = (const float*)d_in[15];

    // ---- workspace carve (256-B aligned) ----
    char* ws = (char*)d_ws;
    size_t off = 0;
    auto take = [&](size_t bytes) -> char* {
        char* p = ws + off;
        off += (bytes + 255) & ~(size_t)255;
        return p;
    };
    __bf16* wt_hole  = (__bf16*)take((size_t)512 * 512 * 2);
    __bf16* wt_fn    = (__bf16*)take((size_t)4 * 512 * 1024 * 2);
    __bf16* wt_rnn   = (__bf16*)take((size_t)512 * 1024 * 2);   // [W_hh ; W_ih] stacked on K
    __bf16* wt_cmp   = (__bf16*)take((size_t)512 * 1024 * 2);
    __bf16* wt_o1    = (__bf16*)take((size_t)512 * 1024 * 2);
    __bf16* x_bf     = (__bf16*)take((size_t)4096 * 512 * 2);
    float*  xh       = (float*) take((size_t)4096 * 512 * 4);
    __bf16* leaf     = (__bf16*)take((size_t)4096 * 16 * 512 * 2);
    __bf16* h1       = (__bf16*)take((size_t)32768 * 512 * 2);
    __bf16* h2       = (__bf16*)take((size_t)16384 * 512 * 2);
    __bf16* h3       = (__bf16*)take((size_t)8192 * 512 * 2);
    __bf16* rootctx  = (__bf16*)take((size_t)4096 * 1024 * 2);  // [root | ctx] bf16
    __bf16* hsA      = (__bf16*)take((size_t)4096 * 512 * 2);
    __bf16* hsB      = (__bf16*)take((size_t)4096 * 512 * 2);
    float*  ctx      = (float*) take((size_t)4096 * 512 * 4);
    float*  cmp      = (float*) take((size_t)4096 * 512 * 4);
    __bf16* compared = (__bf16*)take((size_t)512 * 1024 * 2);
    float*  o1f      = (float*) take((size_t)512 * 512 * 4);

    // ---- weight prep: transpose to N-major bf16 ----
    auto T = [&](__bf16* dst, int dstLd, const float* src, int K, int N) {
        int n = K * N;
        transpose_f32_to_bf16<<<(n + 255) / 256, 256, 0, stream>>>(dst, dstLd, src, K, N);
    };
    T(wt_hole, 512, W_hole, 512, 512);
    for (int d = 0; d < 4; ++d)
        T(wt_fn + (size_t)d * 512 * 1024, 1024, W_fn + (size_t)d * 1024 * 512, 1024, 512);
    T(wt_rnn,        1024, W_hh, 512, 512);   // K 0..511    <- h_{t-1}
    T(wt_rnn + 512,  1024, W_ih, 512, 512);   // K 512..1023 <- seq[t]
    T(wt_cmp, 1024, W_cmp, 1024, 512);
    T(wt_o1,  1024, W_o1,  1024, 512);

    cast_f32_bf16<<<(4096 * 512 + 255) / 256, 256, 0, stream>>>(x, x_bf, 4096 * 512);

    auto G = [&](const __bf16* Alo, int ldaLo, const __bf16* Ahi, int ldaHi, int kSplit,
                 const __bf16* WT, int K, int M, const float* bias,
                 float* Cf, int ldcf, __bf16* Cb, int ldcb, int op) {
        int tiles = (M / 32) * NTILES;
        gemm_wmma_bf16<<<(tiles + 7) / 8, 256, 0, stream>>>(
            Alo, ldaLo, Ahi, ldaHi, kSplit, WT, K, M / 32, bias, Cf, ldcf, Cb, ldcb, op);
    };

    // 1) xh = x @ W_hole  (fp32 out, no activation)
    G(x_bf, 512, nullptr, 0, 512, wt_hole, 512, 4096, nullptr, xh, 512, nullptr, 0, 0);

    // 2) leaf = relu(xh + emb[leaf_ids] + b_hole)  -> bf16, tree layout (be,l,h)
    leaf_build<<<(4096 * 16 * 512 + 255) / 256, 256, 0, stream>>>(xh, leaf_ids, emb, b_hole, leaf);

    // 3) tree reduce: pairs are contiguous 1024-wide rows at each level
    G(leaf, 1024, nullptr, 0, 1024, wt_fn,                        1024, 32768, b_fn,        nullptr, 0, h1,      512,  1);
    G(h1,   1024, nullptr, 0, 1024, wt_fn + (size_t)1 * 512*1024, 1024, 16384, b_fn + 512,  nullptr, 0, h2,      512,  1);
    G(h2,   1024, nullptr, 0, 1024, wt_fn + (size_t)2 * 512*1024, 1024,  8192, b_fn + 1024, nullptr, 0, h3,      512,  1);
    G(h3,   1024, nullptr, 0, 1024, wt_fn + (size_t)3 * 512*1024, 1024,  4096, b_fn + 1536, nullptr, 0, rootctx, 1024, 1);

    // 4) RNN scan: h_t = tanh(h_{t-1}@W_hh + seq[t]@W_ih + b).  Dual-source K=1024 GEMM.
    zero_u32<<<((4096 * 512 * 2 / 4) + 255) / 256, 256, 0, stream>>>((unsigned*)hsA, 4096 * 512 * 2 / 4);
    const __bf16* prev = hsA;
    __bf16* pp[2] = { hsB, hsA };
    for (int t = 0; t < 16; ++t) {
        bool last = (t == 15);
        __bf16* nxt = last ? (rootctx + 512) : pp[t & 1];
        // A_hi = seq[t]: row be -> leaf[(be*16 + t)*512], i.e. base leaf + t*512, stride 8192
        G(prev, 512, leaf + (size_t)t * 512, 8192, 512,
          wt_rnn, 1024, 4096, b_rnn,
          last ? ctx : nullptr, 512, nxt, last ? 1024 : 512, 2);
        prev = nxt;
    }

    // 5) cmp = relu([root|ctx] @ W_cmp + b_cmp)  (fp32)
    G(rootctx, 1024, nullptr, 0, 1024, wt_cmp, 1024, 4096, b_cmp, cmp, 512, nullptr, 0, 1);

    // 6) compared = max_e concat(cmp, ctx)  -> bf16 (512 x 1024)
    compared_max_kernel<<<(512 * 1024 + 255) / 256, 256, 0, stream>>>(cmp, ctx, compared);

    // 7) o1 = relu(compared @ W_o1 + b_o1)  (fp32 512 x 512)
    G(compared, 1024, nullptr, 0, 1024, wt_o1, 1024, 512, b_o1, o1f, 512, nullptr, 0, 1);

    // 8) out = softplus(o1 @ W_o2 + b_o2)
    final_head<<<64, 256, 0, stream>>>(o1f, W_o2, b_o2, (float*)d_out);
}
